// MeriamAttention_55525337203160
// MI455X (gfx1250) — compile-verified, matrix operand order checked by reading, not run on hardware
//
#include <hip/hip_runtime.h>
#include <hip/hip_bf16.h>
#include <stdint.h>

// MI455X / gfx1250: wave32, WMMA 16x16x32 bf16 -> f32 accumulate,
// double-buffered TDM (tensor_load_to_lds) global->LDS tile staging so the
// tensor DMA for tile i+1 overlaps the WMMA work on tile i.

typedef __bf16 bf16_t;
typedef __attribute__((ext_vector_type(16))) __bf16 v16bf;
typedef __attribute__((ext_vector_type(8)))  float  v8f;
typedef __attribute__((ext_vector_type(4)))  unsigned int u32x4;
typedef __attribute__((ext_vector_type(8)))  int i32x8;
typedef __attribute__((ext_vector_type(4)))  int i32x4;

#define B_ 4
#define S_ 2048
#define D_ 1024

#if defined(__has_builtin)
#if __has_builtin(__builtin_amdgcn_tensor_load_to_lds) && __has_builtin(__builtin_amdgcn_s_wait_tensorcnt)
#define USE_TDM 1
#endif
#endif

__device__ __forceinline__ bf16_t f2bf(float f) {
  union { float f; uint32_t u; } c; c.f = f;
  uint32_t u = c.u;
  u += 0x7fffu + ((u >> 16) & 1u);           // round-to-nearest-even
  union { uint16_t u; bf16_t b; } r; r.u = (uint16_t)(u >> 16);
  return r.b;
}
__device__ __forceinline__ float bf2f(bf16_t b) {
  union { bf16_t b; uint16_t u; } c; c.b = b;
  union { uint32_t u; float f; } r; r.u = ((uint32_t)c.u) << 16;
  return r.f;
}

#ifdef USE_TDM
// Issue a 2-D TDM tile load: global (row-major, 2-byte elems) -> LDS.
// tile_d0 elems along the contiguous dim, tile_d1 rows, row stride = stride0
// elems. LDS padding: pad_amount=1 dword every (1<<(pad_code+1)) dwords.
__device__ __forceinline__ void tdm_load_2d(uint32_t lds_off, const void* gaddr,
                                            uint32_t tensor_d0, uint32_t tensor_d1,
                                            uint64_t stride0,
                                            uint32_t tile_d0, uint32_t tile_d1,
                                            uint32_t pad_code) {
  const uint64_t ga = (uint64_t)gaddr;
  u32x4 g0;
  g0[0] = 1u;                                              // count=1, user mode
  g0[1] = lds_off;                                         // lds_addr (bytes)
  g0[2] = (uint32_t)ga;                                    // global_addr[31:0]
  g0[3] = (uint32_t)((ga >> 32) & 0x1FFFFFFu) | (2u << 30); // addr[56:32], type=2

  i32x8 g1;
  // [17:16] data_size=1 (2B), [20] pad_enable, [24:22] pad_interval, [31:25] pad_amount=0 (1 dw)
  g1[0] = (int)((1u << 16) | (1u << 20) | (pad_code << 22));
  g1[1] = (int)((tensor_d0 & 0xFFFFu) << 16);              // [63:48] = td0 lo16
  g1[2] = (int)((tensor_d0 >> 16) | ((tensor_d1 & 0xFFFFu) << 16)); // td0 hi | td1 lo
  g1[3] = (int)((tensor_d1 >> 16) | (tile_d0 << 16));      // td1 hi | tile_dim0
  g1[4] = (int)(tile_d1 & 0xFFFFu);                        // tile_dim1, tile_dim2=0
  g1[5] = (int)(uint32_t)(stride0 & 0xFFFFFFFFu);          // dim0_stride[31:0]
  g1[6] = (int)(uint32_t)((stride0 >> 32) & 0xFFFFu);      // dim0_stride[47:32]
  g1[7] = 0;

  const i32x4 z4 = {0, 0, 0, 0};
#if __clang_major__ >= 23
  const i32x8 z8 = {0, 0, 0, 0, 0, 0, 0, 0};
  __builtin_amdgcn_tensor_load_to_lds(g0, g1, z4, z4, z8, 0);
#else
  __builtin_amdgcn_tensor_load_to_lds(g0, g1, z4, z4, 0);
#endif
}
#endif // USE_TDM

// ---------------------------------------------------------------------------
// f32 -> bf16 conversion
// ---------------------------------------------------------------------------
__global__ void __launch_bounds__(256)
convert_f32_to_bf16(const float* __restrict__ in, bf16_t* __restrict__ out, int n) {
  int i = blockIdx.x * 256 + threadIdx.x;
  if (i < n) out[i] = f2bf(in[i]);
}

// ---------------------------------------------------------------------------
// Tiled WMMA GEMM, double-buffered TDM staging.
//   A: bf16 [M,K] row-major
//   B: if BT: bf16 [N,K] row-major (C = A @ B^T), else bf16 [K,N] row-major
//   C: scale*(A@B) + bias  -> f32 (OUT_F32) or bf16
// Block tile 128x128, K-step 32, 256 threads = 8 waves (2x4 wave grid),
// each wave computes 64x32 = 4x2 WMMA 16x16 accumulators.
// LDS layouts (per buffer): As [128 rows][34 elems] (+1 dword pad / 16 dwords)
//              Bs: BT -> [128][34] ; !BT -> [32][130] (+1 dword pad / 64 dwords)
// ---------------------------------------------------------------------------
template <bool BT, bool HAS_BIAS, bool OUT_F32>
__global__ void __launch_bounds__(256)
wmma_gemm(const bf16_t* __restrict__ A, long long aStrideZ,
          const bf16_t* __restrict__ Bm, long long bStrideZ,
          void* __restrict__ C, long long cStrideZ,
          const float* __restrict__ bias,
          int M, int N, int K, float scale) {
  constexpr int BMt = 128, BNt = 128, BKt = 32;
  constexpr int LDA  = BKt + 2;                 // 34 elems
  constexpr int LDBT = BKt + 2;                 // 34 elems  (BT: [n][k])
  constexpr int LDBN = BNt + 2;                 // 130 elems (!BT: [k][n])
  constexpr int AE = BMt * LDA;                 // A buffer elems
  constexpr int BE = BT ? (BNt * LDBT) : (BKt * LDBN);
  __shared__ bf16_t As[2 * AE];
  __shared__ bf16_t Bs[2 * BE];

  const int tid  = threadIdx.x;
  const int wave = tid >> 5;
  const int lane = tid & 31;
  const int h    = lane >> 4;                   // lane half (K-octet select)
  const int l16  = lane & 15;

  const int bm = blockIdx.x * BMt;
  const int bn = blockIdx.y * BNt;
  const long long z = blockIdx.z;

  const bf16_t* Ap = A  + z * aStrideZ;
  const bf16_t* Bp = Bm + z * bStrideZ;

  const int wm = (wave >> 2) * 64;              // wave row offset: 0 / 64
  const int wn = (wave & 3) * 32;               // wave col offset: 0/32/64/96

  const int ksteps = K / BKt;

  // Stage K-tile `step` into LDS buffer step&1.
  auto stage = [&](int step) {
    const int buf = step & 1;
    bf16_t* Asb = As + buf * AE;
    bf16_t* Bsb = Bs + buf * BE;
    const int k0 = step * BKt;
#ifdef USE_TDM
    if (wave == 0) {
      tdm_load_2d((uint32_t)(uintptr_t)Asb, Ap + (long long)bm * K + k0,
                  (uint32_t)K, (uint32_t)M, (uint64_t)K, BKt, BMt, /*16 dw*/ 3u);
      if (BT) {                                 // B tile [n][k]: BNt x BKt, stride K
        tdm_load_2d((uint32_t)(uintptr_t)Bsb, Bp + (long long)bn * K + k0,
                    (uint32_t)K, (uint32_t)N, (uint64_t)K, BKt, BNt, /*16 dw*/ 3u);
      } else {                                  // B tile [k][n]: BKt x BNt, stride N
        tdm_load_2d((uint32_t)(uintptr_t)Bsb, Bp + (long long)k0 * N + bn,
                    (uint32_t)N, (uint32_t)K, (uint64_t)N, BNt, BKt, /*64 dw*/ 5u);
      }
    }
#else
    #pragma unroll
    for (int i = tid; i < 512; i += 256) {
      const int r = i >> 2, cq = i & 3;
      const uint4 v = *reinterpret_cast<const uint4*>(Ap + (long long)(bm + r) * K + k0 + cq * 8);
      uint32_t* dst = reinterpret_cast<uint32_t*>(&Asb[r * LDA + cq * 8]);
      dst[0] = v.x; dst[1] = v.y; dst[2] = v.z; dst[3] = v.w;
    }
    if (BT) {
      #pragma unroll
      for (int i = tid; i < 512; i += 256) {
        const int r = i >> 2, cq = i & 3;
        const uint4 v = *reinterpret_cast<const uint4*>(Bp + (long long)(bn + r) * K + k0 + cq * 8);
        uint32_t* dst = reinterpret_cast<uint32_t*>(&Bsb[r * LDBT + cq * 8]);
        dst[0] = v.x; dst[1] = v.y; dst[2] = v.z; dst[3] = v.w;
      }
    } else {
      #pragma unroll
      for (int i = tid; i < 512; i += 256) {
        const int kk = i >> 4, nq = i & 15;
        const uint4 v = *reinterpret_cast<const uint4*>(Bp + (long long)(k0 + kk) * N + bn + nq * 8);
        uint32_t* dst = reinterpret_cast<uint32_t*>(&Bsb[kk * LDBN + nq * 8]);
        dst[0] = v.x; dst[1] = v.y; dst[2] = v.z; dst[3] = v.w;
      }
    }
#endif
  };

  v8f acc[4][2] = {};

  // Prologue: stage tile 0 and make it visible.
  stage(0);
#ifdef USE_TDM
  if (wave == 0) __builtin_amdgcn_s_wait_tensorcnt(0);
#endif
  __syncthreads();

  for (int step = 0; step < ksteps; ++step) {
    // Kick off the DMA for the next tile; it overlaps this step's WMMAs.
    if (step + 1 < ksteps) stage(step + 1);

    const bf16_t* Asb = As + (step & 1) * AE;
    const bf16_t* Bsb = Bs + (step & 1) * BE;

    // --- build fragments (documented 16-bit A/B WMMA layout):
    // lane: row/col = lane%16; half h selects K octet; VGPR v holds the K pair
    // at kb = (v<4 ? 0 : 16) + h*8 + (v%4)*2
    union Frag { v16bf v; uint32_t u[8]; };
    Frag afr[4], bfr[2];
    #pragma unroll
    for (int mi = 0; mi < 4; ++mi) {
      const int row = wm + mi * 16 + l16;
      #pragma unroll
      for (int v = 0; v < 8; ++v) {
        const int kb = ((v & 4) << 2) + h * 8 + (v & 3) * 2;
        afr[mi].u[v] = *reinterpret_cast<const uint32_t*>(&Asb[row * LDA + kb]);
      }
    }
    #pragma unroll
    for (int ni = 0; ni < 2; ++ni) {
      const int col = wn + ni * 16 + l16;
      #pragma unroll
      for (int v = 0; v < 8; ++v) {
        const int kb = ((v & 4) << 2) + h * 8 + (v & 3) * 2;
        if (BT) {
          bfr[ni].u[v] = *reinterpret_cast<const uint32_t*>(&Bsb[col * LDBT + kb]);
        } else {
          const uint16_t* bs = reinterpret_cast<const uint16_t*>(Bsb);
          const uint32_t lo = bs[(kb + 0) * LDBN + col];
          const uint32_t hi = bs[(kb + 1) * LDBN + col];
          bfr[ni].u[v] = lo | (hi << 16);
        }
      }
    }

    // --- 8 WMMAs per wave per K step ---
    #pragma unroll
    for (int mi = 0; mi < 4; ++mi)
      #pragma unroll
      for (int ni = 0; ni < 2; ++ni)
        acc[mi][ni] = __builtin_amdgcn_wmma_f32_16x16x32_bf16(
            false, afr[mi].v, false, bfr[ni].v, (short)0, acc[mi][ni], false, false);

    // All waves done reading buffer step&1; next tile must be resident before
    // the next iteration's fragment gathers.
    __syncthreads();
#ifdef USE_TDM
    if (wave == 0) __builtin_amdgcn_s_wait_tensorcnt(0);
    __syncthreads();
#endif
  }

  // --- writeback: C/D layout: lane l16 = N, VGPR j -> M = j + 8*h ---
  #pragma unroll
  for (int mi = 0; mi < 4; ++mi) {
    #pragma unroll
    for (int ni = 0; ni < 2; ++ni) {
      const int col = bn + wn + ni * 16 + l16;
      const float bv = HAS_BIAS ? bias[col] : 0.0f;
      #pragma unroll
      for (int j = 0; j < 8; ++j) {
        const int row = bm + wm + mi * 16 + j + 8 * h;
        const float val = acc[mi][ni][j] * scale + bv;
        if constexpr (OUT_F32)
          reinterpret_cast<float*>(C)[z * cStrideZ + (long long)row * N + col] = val;
        else
          reinterpret_cast<bf16_t*>(C)[z * cStrideZ + (long long)row * N + col] = f2bf(val);
      }
    }
  }
}

// ---------------------------------------------------------------------------
// In-place row softmax over bf16 scores (f32 math), one block per row.
// ---------------------------------------------------------------------------
__global__ void __launch_bounds__(256)
softmax_rows(bf16_t* __restrict__ sc, int n) {
  __shared__ float buf[S_];
  __shared__ float red[8];
  __shared__ float sMax, sInv;
  const long long row = blockIdx.x;
  bf16_t* p = sc + row * (long long)n;
  const int tid = threadIdx.x, lane = tid & 31, wave = tid >> 5;

  float lmax = -3.4e38f;
  for (int i = tid; i < n; i += 256) {
    const float v = bf2f(p[i]);
    buf[i] = v;
    lmax = fmaxf(lmax, v);
  }
  #pragma unroll
  for (int o = 16; o; o >>= 1) lmax = fmaxf(lmax, __shfl_xor(lmax, o, 32));
  if (lane == 0) red[wave] = lmax;
  __syncthreads();
  if (tid == 0) {
    float m = red[0];
    for (int i = 1; i < 8; ++i) m = fmaxf(m, red[i]);
    sMax = m;
  }
  __syncthreads();
  const float m = sMax;

  float lsum = 0.0f;
  for (int i = tid; i < n; i += 256) {
    const float e = __expf(buf[i] - m);
    buf[i] = e;
    lsum += e;
  }
  #pragma unroll
  for (int o = 16; o; o >>= 1) lsum += __shfl_xor(lsum, o, 32);
  if (lane == 0) red[wave] = lsum;
  __syncthreads();
  if (tid == 0) {
    float s = 0.0f;
    for (int i = 0; i < 8; ++i) s += red[i];
    sInv = 1.0f / s;
  }
  __syncthreads();
  const float inv = sInv;
  for (int i = tid; i < n; i += 256) p[i] = f2bf(buf[i] * inv);
}

// ---------------------------------------------------------------------------
// Host-side launcher
// ---------------------------------------------------------------------------
extern "C" void kernel_launch(void* const* d_in, const int* in_sizes, int n_in,
                              void* d_out, int out_size, void* d_ws, size_t ws_size,
                              hipStream_t stream) {
  (void)in_sizes; (void)n_in; (void)out_size; (void)ws_size;
  const float* x  = (const float*)d_in[0];
  const float* Wq = (const float*)d_in[1];
  const float* bq = (const float*)d_in[2];
  const float* Wk = (const float*)d_in[3];
  const float* bk = (const float*)d_in[4];
  const float* Wv = (const float*)d_in[5];
  const float* bv = (const float*)d_in[6];
  float* out = (float*)d_out;

  const long long nx = (long long)B_ * S_ * D_;   // 8,388,608
  const long long nw = (long long)D_ * D_;        // 1,048,576
  const long long sd = (long long)S_ * D_;
  const long long ss = (long long)S_ * S_;

  // Workspace carve-up (~107 MB total)
  char* ws = (char*)d_ws;
  size_t off = 0;
  auto carve = [&](size_t bytes) {
    char* p = ws + off;
    off = (off + bytes + 255) & ~(size_t)255;
    return p;
  };
  bf16_t* xbf = (bf16_t*)carve(nx * 2);
  bf16_t* wqb = (bf16_t*)carve(nw * 2);
  bf16_t* wkb = (bf16_t*)carve(nw * 2);
  bf16_t* wvb = (bf16_t*)carve(nw * 2);
  bf16_t* qbf = (bf16_t*)carve(nx * 2);
  bf16_t* kbf = (bf16_t*)carve(nx * 2);
  bf16_t* vbf = (bf16_t*)carve(nx * 2);
  bf16_t* sbf = (bf16_t*)carve((long long)B_ * ss * 2);

  // 1) precision convert
  convert_f32_to_bf16<<<(int)(nx / 256), 256, 0, stream>>>(x,  xbf, (int)nx);
  convert_f32_to_bf16<<<(int)(nw / 256), 256, 0, stream>>>(Wq, wqb, (int)nw);
  convert_f32_to_bf16<<<(int)(nw / 256), 256, 0, stream>>>(Wk, wkb, (int)nw);
  convert_f32_to_bf16<<<(int)(nw / 256), 256, 0, stream>>>(Wv, wvb, (int)nw);

  // 2) QKV projections: [8192,1024] x [1024,1024] + bias -> bf16
  dim3 gq(8192 / 128, 1024 / 128, 1);
  wmma_gemm<false, true, false><<<gq, 256, 0, stream>>>(
      xbf, 0, wqb, 0, (void*)qbf, 0, bq, 8192, 1024, 1024, 1.0f);
  wmma_gemm<false, true, false><<<gq, 256, 0, stream>>>(
      xbf, 0, wkb, 0, (void*)kbf, 0, bk, 8192, 1024, 1024, 1.0f);
  wmma_gemm<false, true, false><<<gq, 256, 0, stream>>>(
      xbf, 0, wvb, 0, (void*)vbf, 0, bv, 8192, 1024, 1024, 1.0f);

  // 3) scores = scale * Q @ K^T  (per batch via grid.z)
  dim3 gs(2048 / 128, 2048 / 128, B_);
  wmma_gemm<true, false, false><<<gs, 256, 0, stream>>>(
      qbf, sd, kbf, sd, (void*)sbf, ss, nullptr, 2048, 2048, 1024, 0.03125f);

  // 4) softmax rows (in place)
  softmax_rows<<<B_ * S_, 256, 0, stream>>>(sbf, S_);

  // 5) out = P @ V -> f32
  dim3 gp(2048 / 128, 1024 / 128, B_);
  wmma_gemm<false, false, true><<<gp, 256, 0, stream>>>(
      sbf, ss, vbf, sd, (void*)out, sd, nullptr, 2048, 1024, 2048, 1.0f);
}
